// MultiHeadAttention_7017976561971
// MI455X (gfx1250) — matrix operand order repaired and checked by reading, not compiled, not run on hardware
//
#include <hip/hip_runtime.h>
#include <hip/hip_bf16.h>

typedef _Float16 v8h  __attribute__((ext_vector_type(8)));
typedef _Float16 v16h __attribute__((ext_vector_type(16)));
typedef float    v8f  __attribute__((ext_vector_type(8)));

#define D_EMB 768
#define NH    12
#define HD    64
#define BB    4
#define SS    2048
#define NTOK  (BB * SS)        /* 8192 */
#define KT32  (D_EMB / 32)     /* 24 K-tiles of 32 */

static __device__ __forceinline__ v16h hcat(v8h lo, v8h hi) {
  return __builtin_shufflevector(lo, hi, 0,1,2,3,4,5,6,7,8,9,10,11,12,13,14,15);
}
static __device__ __forceinline__ v8f wmma16(v16h a, v16h b, v8f c) {
  // v_wmma_f32_16x16x32_f16
  return __builtin_amdgcn_wmma_f32_16x16x32_f16(false, a, false, b, (short)0, c,
                                                false, false);
}

// Async global->LDS copy of one 16-byte beat (gfx1250, ASYNCcnt-tracked).
static __device__ __forceinline__ void async_g2l_b128(unsigned lds_byte_addr,
                                                      unsigned long long gaddr) {
  asm volatile("global_load_async_to_lds_b128 %0, %1, off"
               :: "v"(lds_byte_addr), "v"(gaddr) : "memory");
}
static __device__ __forceinline__ void wait_async0() {
  asm volatile("s_wait_asynccnt 0x0" ::: "memory");
}

// ---------------------------------------------------------------------------
// Pack fp32 weights into B-fragment-native f16 blobs.
// Blob layout per (kt,nt) 32x16 tile: [lane 0..31][half 0..15] contiguous,
//   lane = (n%16) + 16*((k%32)/16), half = (k%32)%16   (contiguous k order, same as A)
// flat = ((nt*KT32 + kt)*32 + lane)*16 + half
// => 96KB panel for 4 consecutive nt is one contiguous region (async-stageable)
// ---------------------------------------------------------------------------
__global__ void pack_qkv_k(const float* __restrict__ Wq, const float* __restrict__ Wk,
                           const float* __restrict__ Wv, _Float16* __restrict__ Wp) {
  int idx = blockIdx.x * 256 + threadIdx.x;
  if (idx >= 3 * D_EMB * D_EMB) return;
  int n = idx / D_EMB;           // 0..2303 (mtx-major cols)
  int k = idx % D_EMB;           // 0..767
  int mtx = n / D_EMB, c = n % D_EMB;
  int h = c >> 6, e = c & 63;
  const float* W = (mtx == 0) ? Wq : (mtx == 1) ? Wk : Wv;
  float v = W[(h * D_EMB + k) * HD + e];
  int nt = n >> 4, nn = n & 15, kt = k >> 5, kk = k & 31;
  int lane = nn + 16 * (kk >> 4), half = kk & 15;
  Wp[(((size_t)nt * KT32 + kt) * 32 + lane) * 16 + half] = (_Float16)v;
}

__global__ void pack_wo_k(const float* __restrict__ Wo, _Float16* __restrict__ Wp) {
  int idx = blockIdx.x * 256 + threadIdx.x;
  if (idx >= D_EMB * D_EMB) return;
  int n = idx / D_EMB, k = idx % D_EMB;
  float v = Wo[k * D_EMB + n];
  int nt = n >> 4, nn = n & 15, kt = k >> 5, kk = k & 31;
  int lane = nn + 16 * (kk >> 4), half = kk & 15;
  Wp[(((size_t)nt * KT32 + kt) * 32 + lane) * 16 + half] = (_Float16)v;
}

__global__ void cvt_hidden_k(const float* __restrict__ X, _Float16* __restrict__ Xh) {
  int idx = blockIdx.x * 256 + threadIdx.x;
  if (idx < NTOK * D_EMB) Xh[idx] = (_Float16)X[idx];
}

// ---------------------------------------------------------------------------
// GEMM body shared by both projection kernels:
//   - async-stage the block's contiguous 96KB packed B panel into LDS
//   - each wave computes a 16x64 tile; B fragments come from LDS
//   - software-pipelined: next step's A (global) + 4 B fragments (LDS) are
//     loaded before the current step's 4 WMMAs issue
// ---------------------------------------------------------------------------
#define GEMM_PROLOG(APTR, WPPTR)                                                 \
  __shared__ __align__(128) _Float16 Bs[4 * KT32 * 32 * 16]; /* 96 KB */         \
  const int lane = threadIdx.x & 31;                                             \
  const int w    = threadIdx.x >> 5;                                             \
  const int m0   = blockIdx.x * 128 + w * 16;                                    \
  const int n0   = blockIdx.y * 64;                                              \
  const int nn   = lane & 15, hi = lane >> 4;                                    \
  {                                                                              \
    const unsigned lds0 = (unsigned)(uintptr_t)&Bs[0];                           \
    const unsigned long long g0 =                                                \
        (unsigned long long)(uintptr_t)((WPPTR) + (size_t)(n0 >> 4) * KT32 * 512);\
    _Pragma("unroll")                                                            \
    for (int r = 0; r < 24; ++r) {                                               \
      unsigned boff = (unsigned)(r * 256 + threadIdx.x) * 16u;                   \
      async_g2l_b128(lds0 + boff, g0 + boff);                                    \
    }                                                                            \
    wait_async0();                                                               \
  }                                                                              \
  __syncthreads();                                                               \
  v8f acc[4] = {};                                                               \
  const _Float16* arow = (APTR) + (size_t)(m0 + nn) * D_EMB;                     \
  const _Float16* bbase = Bs + (size_t)lane * 16;                                \
  v16h afr = *(const v16h*)(arow + 16 * hi);                                     \
  v16h bfr[4];                                                                   \
  _Pragma("unroll")                                                              \
  for (int j = 0; j < 4; ++j)                                                    \
    bfr[j] = *(const v16h*)(bbase + (size_t)j * KT32 * 512);                     \
  _Pragma("unroll 4")                                                            \
  for (int kt = 0; kt < KT32 - 1; ++kt) {                                        \
    const int k1 = (kt + 1) * 32;                                                \
    __builtin_prefetch(arow + k1 + 224, 0, 1);                                   \
    v16h an = *(const v16h*)(arow + k1 + 16 * hi);                               \
    v16h bn[4];                                                                  \
    _Pragma("unroll")                                                            \
    for (int j = 0; j < 4; ++j)                                                  \
      bn[j] = *(const v16h*)(bbase + ((size_t)j * KT32 + (kt + 1)) * 512);       \
    _Pragma("unroll")                                                            \
    for (int j = 0; j < 4; ++j) acc[j] = wmma16(afr, bfr[j], acc[j]);            \
    afr = an;                                                                    \
    _Pragma("unroll")                                                            \
    for (int j = 0; j < 4; ++j) bfr[j] = bn[j];                                  \
  }                                                                              \
  _Pragma("unroll")                                                              \
  for (int j = 0; j < 4; ++j) acc[j] = wmma16(afr, bfr[j], acc[j]);

// Fused QKV GEMM: Xh[8192,768] x Wp[768,2304] (+bias).
// Q gets the 1/sqrt(HD)=0.125 scale folded in.
// Q,K stored [BH][S][64]; V stored transposed [BH][64][S].
__global__ void gemm_qkv_k(const _Float16* __restrict__ Xh, const _Float16* __restrict__ Wp,
                           const float* __restrict__ bq, const float* __restrict__ bk,
                           const float* __restrict__ bv,
                           _Float16* __restrict__ Qp, _Float16* __restrict__ Kp,
                           _Float16* __restrict__ Vt) {
  GEMM_PROLOG(Xh, Wp)
  const int b2 = m0 >> 11;
  const int sb = m0 & (SS - 1);
#pragma unroll
  for (int j = 0; j < 4; ++j) {
    int n = n0 + j * 16 + nn;
    int mtx = n / D_EMB, c = n % D_EMB;
    int h = c >> 6, e = c & 63;
    size_t bh = (size_t)b2 * NH + h;
    if (mtx == 2) {  // V: 8 consecutive s per lane -> one b128 store
      v8h vv;
#pragma unroll
      for (int r = 0; r < 8; ++r) vv[r] = (_Float16)(acc[j][r] + bv[c]);
      *(v8h*)&Vt[(bh * HD + e) * SS + sb + 8 * hi] = vv;
    } else if (mtx == 0) {
#pragma unroll
      for (int r = 0; r < 8; ++r) {
        int s = sb + r + 8 * hi;
        Qp[(bh * SS + s) * HD + e] = (_Float16)((acc[j][r] + bq[c]) * 0.125f);
      }
    } else {
#pragma unroll
      for (int r = 0; r < 8; ++r) {
        int s = sb + r + 8 * hi;
        Kp[(bh * SS + s) * HD + e] = (_Float16)(acc[j][r] + bk[c]);
      }
    }
  }
}

// Output projection: Xc[8192,768] x Wo[768,768] + bo -> fp32 out
__global__ void gemm_out_k(const _Float16* __restrict__ Xc, const _Float16* __restrict__ Wp,
                           const float* __restrict__ bo, float* __restrict__ out) {
  GEMM_PROLOG(Xc, Wp)
#pragma unroll
  for (int j = 0; j < 4; ++j) {
#pragma unroll
    for (int r = 0; r < 8; ++r) {
      int m = m0 + r + 8 * hi;
      int n = n0 + j * 16 + nn;
      out[(size_t)m * D_EMB + n] = acc[j][r] + bo[n];
    }
  }
}

// ---------------------------------------------------------------------------
// Flash attention: one wave = 16 query rows, streams 2048 keys in chunks of 32.
// All 8 K/V fragments of a chunk are loaded up front so global latency
// overlaps the softmax VALU/transcendental work.
// ---------------------------------------------------------------------------
__global__ void attn_k(const _Float16* __restrict__ Qp, const _Float16* __restrict__ Kp,
                       const _Float16* __restrict__ Vt, _Float16* __restrict__ Xc) {
  __shared__ __align__(64) _Float16 lp[8][16][32];   // per-wave P tile (16x32)
  const int lane = threadIdx.x & 31;
  const int w    = threadIdx.x >> 5;
  const int bh   = blockIdx.x;               // 0..47
  const int b    = bh / NH, h = bh % NH;
  const int s0   = blockIdx.y * 128 + w * 16;
  const int nn   = lane & 15, hi = lane >> 4;

  const _Float16* qb = Qp + ((size_t)bh * SS + s0) * HD;
  const _Float16* kb = Kp + (size_t)bh * SS * HD;
  const _Float16* vb = Vt + (size_t)bh * HD * SS;

  const _Float16* qr = qb + nn * HD;
  v16h qa0 = *(const v16h*)(qr + 16 * hi);
  v16h qa1 = *(const v16h*)(qr + 32 + 16 * hi);

  v8f o0 = {}, o1 = {}, o2 = {}, o3 = {};
  float mrow[8], lrow[8];
#pragma unroll
  for (int r = 0; r < 8; ++r) { mrow[r] = -1e30f; lrow[r] = 0.f; }

  for (int t0 = 0; t0 < SS; t0 += 32) {
    // --- load ALL chunk fragments up front (4x K, 4x V) ---
    const _Float16* kp0 = kb + (size_t)(t0 + nn) * HD + 16 * hi;
    const _Float16* kp1 = kb + (size_t)(t0 + 16 + nn) * HD + 16 * hi;
    const _Float16* vp  = vb + t0 + 16 * hi + (size_t)nn * SS;
    v16h kf0 = *(const v16h*)(kp0);
    v16h kf1 = *(const v16h*)(kp0 + 32);
    v16h kf2 = *(const v16h*)(kp1);
    v16h kf3 = *(const v16h*)(kp1 + 32);
    v16h vf0 = *(const v16h*)(vp + 0 * 16 * SS);
    v16h vf1 = *(const v16h*)(vp + 1 * 16 * SS);
    v16h vf2 = *(const v16h*)(vp + 2 * 16 * SS);
    v16h vf3 = *(const v16h*)(vp + 3 * 16 * SS);
    __builtin_prefetch(kp0 + 32 * HD, 0, 1);       // next chunk K
    __builtin_prefetch(vp + 32, 0, 1);             // next chunk V

    // --- scores: two 16x16 tiles over keys [t0, t0+32) ---
    v8f sf0 = {}, sf1 = {};
    sf0 = wmma16(qa0, kf0, sf0);
    sf0 = wmma16(qa1, kf1, sf0);
    sf1 = wmma16(qa0, kf2, sf1);
    sf1 = wmma16(qa1, kf3, sf1);

    // --- online softmax update (row = r + 8*hi, cols across 16-lane group) ---
    float sc[8];
#pragma unroll
    for (int r = 0; r < 8; ++r) {
      float mx = fmaxf(sf0[r], sf1[r]);
      mx = fmaxf(mx, __shfl_xor(mx, 1, 16));
      mx = fmaxf(mx, __shfl_xor(mx, 2, 16));
      mx = fmaxf(mx, __shfl_xor(mx, 4, 16));
      mx = fmaxf(mx, __shfl_xor(mx, 8, 16));
      float mn = fmaxf(mrow[r], mx);
      sc[r] = __expf(mrow[r] - mn);
      float p0 = __expf(sf0[r] - mn);
      float p1 = __expf(sf1[r] - mn);
      lp[w][r + 8 * hi][nn]      = (_Float16)p0;
      lp[w][r + 8 * hi][16 + nn] = (_Float16)p1;
      float rs = p0 + p1;
      rs += __shfl_xor(rs, 1, 16);
      rs += __shfl_xor(rs, 2, 16);
      rs += __shfl_xor(rs, 4, 16);
      rs += __shfl_xor(rs, 8, 16);
      lrow[r] = lrow[r] * sc[r] + rs;
      mrow[r] = mn;
    }
    __builtin_amdgcn_wave_barrier();   // DS ops are in-order per wave; fence codegen
    // --- P C-layout -> A-layout via LDS ---
    v16h pa = *(const v16h*)(&lp[w][nn][16 * hi]);
    __builtin_amdgcn_wave_barrier();
    // --- rescale accumulators, then O += P * V ---
#pragma unroll
    for (int r = 0; r < 8; ++r) {
      o0[r] *= sc[r]; o1[r] *= sc[r]; o2[r] *= sc[r]; o3[r] *= sc[r];
    }
    o0 = wmma16(pa, vf0, o0);
    o1 = wmma16(pa, vf1, o1);
    o2 = wmma16(pa, vf2, o2);
    o3 = wmma16(pa, vf3, o3);
  }
  // --- finalize: divide by row sums, write concat tile ---
#pragma unroll
  for (int r = 0; r < 8; ++r) {
    int s = s0 + r + 8 * hi;
    size_t rowb = ((size_t)b * SS + s) * D_EMB + h * HD;
    float inv = 1.0f / lrow[r];
    Xc[rowb + 0 * 16 + nn] = (_Float16)(o0[r] * inv);
    Xc[rowb + 1 * 16 + nn] = (_Float16)(o1[r] * inv);
    Xc[rowb + 2 * 16 + nn] = (_Float16)(o2[r] * inv);
    Xc[rowb + 3 * 16 + nn] = (_Float16)(o3[r] * inv);
  }
}

// ---------------------------------------------------------------------------
extern "C" void kernel_launch(void* const* d_in, const int* in_sizes, int n_in,
                              void* d_out, int out_size, void* d_ws, size_t ws_size,
                              hipStream_t stream) {
  const float* hidden = (const float*)d_in[0];
  const float* Wq = (const float*)d_in[1];  const float* bq = (const float*)d_in[2];
  const float* Wk = (const float*)d_in[3];  const float* bk = (const float*)d_in[4];
  const float* Wv = (const float*)d_in[5];  const float* bv = (const float*)d_in[6];
  const float* Wo = (const float*)d_in[7];  const float* bo = (const float*)d_in[8];
  float* out = (float*)d_out;

  char* ws = (char*)d_ws;
  const size_t SZ_XH   = (size_t)NTOK * D_EMB * 2;        // 12.6 MB
  const size_t SZ_WQKV = (size_t)3 * D_EMB * D_EMB * 2;   // 3.5 MB
  const size_t SZ_WO   = (size_t)D_EMB * D_EMB * 2;       // 1.2 MB
  size_t off = 0;
  _Float16* Xh    = (_Float16*)(ws + off); off += SZ_XH;
  _Float16* Wqkvp = (_Float16*)(ws + off); off += SZ_WQKV;
  _Float16* Wop   = (_Float16*)(ws + off); off += SZ_WO;
  _Float16* Qp    = (_Float16*)(ws + off); off += SZ_XH;
  _Float16* Kp    = (_Float16*)(ws + off); off += SZ_XH;
  _Float16* Vt    = (_Float16*)(ws + off); off += SZ_XH;
  _Float16* Xc    = (_Float16*)(ws + off); off += SZ_XH;

  cvt_hidden_k<<<(NTOK * D_EMB + 255) / 256, 256, 0, stream>>>(hidden, Xh);
  pack_qkv_k<<<(3 * D_EMB * D_EMB + 255) / 256, 256, 0, stream>>>(Wq, Wk, Wv, Wqkvp);
  pack_wo_k<<<(D_EMB * D_EMB + 255) / 256, 256, 0, stream>>>(Wo, Wop);
  gemm_qkv_k<<<dim3(NTOK / 128, (3 * D_EMB) / 64), 256, 0, stream>>>(Xh, Wqkvp, bq, bk, bv,
                                                                     Qp, Kp, Vt);
  attn_k<<<dim3(BB * NH, SS / 128), 256, 0, stream>>>(Qp, Kp, Vt, Xc);
  gemm_out_k<<<dim3(NTOK / 128, D_EMB / 64), 256, 0, stream>>>(Xc, Wop, bo, out);
}
